// RandomSparseTransformer_47090021433649
// MI455X (gfx1250) — matrix operand order repaired and checked by reading, not compiled, not run on hardware
//
#include <hip/hip_runtime.h>

#define B_  16
#define L_  512
#define H_  768
#define FF_ 3072
#define NL_ 6
#define NH_ 12
#define HD_ 64
#define M_  (B_*L_)      // 8192 tokens
#define NSKIP_ 15
#define NC_ 2

typedef __attribute__((ext_vector_type(16))) __bf16       v16bf;
typedef __attribute__((ext_vector_type(8)))  float        v8f;
typedef __attribute__((ext_vector_type(8)))  unsigned int v8u;
typedef __attribute__((ext_vector_type(4)))  unsigned int u32x4;
typedef __attribute__((ext_vector_type(8)))  int          i32x8;
typedef __attribute__((ext_vector_type(4)))  int          i32x4;

#if defined(__has_builtin)
#if __has_builtin(__builtin_amdgcn_tensor_load_to_lds)
#define HAVE_TDM 1
#endif
#endif
#ifndef HAVE_TDM
#define HAVE_TDM 0
#endif

#define WMMA_BF16(A, Bm, C) \
  __builtin_amdgcn_wmma_f32_16x16x32_bf16(false, (A), false, (Bm), (short)0, (C), false, false)

__device__ __forceinline__ unsigned short f2bf(float f) {
  unsigned int u = __float_as_uint(f);
  u = (u + 0x7FFFu + ((u >> 16) & 1u)) >> 16;   // round-to-nearest-even
  return (unsigned short)u;
}

// Build a v16bf fragment from two 16-byte contiguous runs of bf16.
__device__ __forceinline__ v16bf frag_from2(const unsigned short* p0,
                                            const unsigned short* p1) {
  uint4 a = *(const uint4*)p0;
  uint4 b = *(const uint4*)p1;
  v8u r;
  r[0] = a.x; r[1] = a.y; r[2] = a.z; r[3] = a.w;
  r[4] = b.x; r[5] = b.y; r[6] = b.z; r[7] = b.w;
  return __builtin_bit_cast(v16bf, r);
}
// One 32-byte contiguous run of 16 bf16.
__device__ __forceinline__ v16bf frag_from32(const unsigned short* p) {
  return frag_from2(p, p + 8);
}

#if HAVE_TDM
// TDM: DMA a 2D bf16 tile [rows=64 (dim1), cols=32 (dim0)] from global (row stride
// Kdim elements) into LDS at lds_off, rows packed contiguously (64B apart).
// D# built per CDNA5 ISA 8.3/8.4: group0 = {count, lds_addr, global_addr, type=2},
// group1 = {data_size=2B, tensor dims, tile dims, dim0 stride}.
// This toolchain's builtin takes 6 args: (g0:u32x4, g1:i32x8, g2:i32x4, g3:i32x4,
// extra:i32x8, cpol:i32).
__device__ __forceinline__ void tdm_load_tile(const unsigned short* gsrc,
                                              unsigned lds_off, int Kdim, int Ndim) {
  unsigned long long ga = (unsigned long long)(uintptr_t)gsrc;
  u32x4 g0;
  g0[0] = 1u;                                                 // count = 1 valid D#
  g0[1] = lds_off;                                            // lds_addr (bytes)
  g0[2] = (unsigned)ga;                                       // global_addr[31:0]
  g0[3] = (unsigned)((ga >> 32) & 0x01FFFFFFu) | 0x80000000u; // addr[56:32] | type=2
  i32x8 g1;
  g1[0] = 0x00010000;                                         // data_size=1 (2 bytes)
  g1[1] = (int)(((unsigned)Kdim & 0xFFFFu) << 16);            // tensor_dim0[15:0]
  g1[2] = (int)((((unsigned)Kdim >> 16) & 0xFFFFu) |
                (((unsigned)Ndim & 0xFFFFu) << 16));          // dim0 hi | dim1 lo
  g1[3] = (int)((((unsigned)Ndim >> 16) & 0xFFFFu) |
                (32u << 16));                                  // dim1 hi | tile_dim0=32
  g1[4] = 64;                                                 // tile_dim1=64, tile_dim2=0
  g1[5] = Kdim;                                               // tensor_dim0_stride[31:0]
  g1[6] = 0;
  g1[7] = 0;
  i32x4 z4 = {0, 0, 0, 0};
  i32x8 z8 = {0, 0, 0, 0, 0, 0, 0, 0};
  __builtin_amdgcn_tensor_load_to_lds(g0, g1, z4, z4, z8, 0);
}

__device__ __forceinline__ void tdm_wait0() {
#if __has_builtin(__builtin_amdgcn_s_wait_tensorcnt)
  __builtin_amdgcn_s_wait_tensorcnt(0);
#else
  asm volatile("s_wait_tensorcnt 0x0" ::: "memory");
#endif
}
#endif  // HAVE_TDM

// ---------------- elementwise prep kernels ----------------

__global__ void k_wprep(const float* __restrict__ w, const float* __restrict__ msk,
                        unsigned short* __restrict__ dst, int n) {
  int i = blockIdx.x * 256 + threadIdx.x;
  if (i < n) {
    float v = w[i];
    if (msk) v *= msk[i];
    dst[i] = f2bf(v);
  }
}

__global__ void k_embed(const int* __restrict__ ids, const float* __restrict__ te,
                        const float* __restrict__ pe, float* __restrict__ of,
                        unsigned short* __restrict__ ob) {
  int i = blockIdx.x * 256 + threadIdx.x;   // over M_*H_ exactly
  int m = i / H_, c = i % H_;
  int tok = ids[m];
  float v = te[(size_t)tok * H_ + c] + pe[(size_t)(m % L_) * H_ + c];
  of[i] = v;
  ob[i] = f2bf(v);
}

__global__ void k_f2bf(const float* __restrict__ x, unsigned short* __restrict__ o, int n) {
  int i = blockIdx.x * 256 + threadIdx.x;
  if (i < n) o[i] = f2bf(x[i]);
}

// [B,L,H] bf16 -> [B,NH,HD,L] bf16 (so p@v B-fragments are contiguous loads)
__global__ void k_transpose_v(const unsigned short* __restrict__ src,
                              unsigned short* __restrict__ dst) {
  int i = blockIdx.x * 256 + threadIdx.x;   // over M_*H_ exactly
  int m = i / H_, c = i % H_;
  int b = m / L_, l = m % L_;
  int h = c / HD_, d = c % HD_;
  dst[(((size_t)b * NH_ + h) * HD_ + d) * L_ + l] = src[i];
}

// ---------------- layernorm (one wave per token) ----------------
__global__ void k_ln(const float* __restrict__ x, const float* __restrict__ g,
                     const float* __restrict__ b, unsigned short* __restrict__ o) {
  int w = threadIdx.x >> 5, lane = threadIdx.x & 31;
  int tok = blockIdx.x * 8 + w;
  const float* xr = x + (size_t)tok * H_;
  float xv[H_ / 32];
  float s = 0.f, sq = 0.f;
#pragma unroll
  for (int j = 0; j < H_ / 32; ++j) {
    float v = xr[lane + j * 32];
    xv[j] = v; s += v; sq += v * v;
  }
#pragma unroll
  for (int off = 16; off; off >>= 1) {
    s  += __shfl_xor(s,  off, 32);
    sq += __shfl_xor(sq, off, 32);
  }
  float mean = s * (1.f / H_);
  float var  = sq * (1.f / H_) - mean * mean;
  float inv  = rsqrtf(var + 1e-5f);
  unsigned short* orow = o + (size_t)tok * H_;
#pragma unroll
  for (int j = 0; j < H_ / 32; ++j) {
    int c = lane + j * 32;
    orow[c] = f2bf((xv[j] - mean) * inv * g[c] + b[c]);
  }
}

// ---------------- generic bf16 WMMA GEMM: Y = X @ W^T (+bias)(+add)(relu) ----------------
// X: [Mdim,Kdim] bf16, W: [Ndim,Kdim] bf16 (pre-masked).
// Block = 8 waves covering a 128x64 tile. The 64x32 weight K-chunk shared by all
// 8 waves is staged in LDS by the Tensor Data Mover (double-buffered, TENSORcnt),
// cutting weight traffic 8x; A fragments stream from global per wave.
__global__ void __launch_bounds__(256)
k_gemm(const unsigned short* __restrict__ X, const unsigned short* __restrict__ W,
       const float* __restrict__ bias, const float* __restrict__ add,
       float* __restrict__ outF, unsigned short* __restrict__ outB,
       int Mdim, int Ndim, int Kdim, int relu) {
  __shared__ __align__(256) unsigned short wtile[2][64][32];
  int wid = threadIdx.x >> 5, lane = threadIdx.x & 31;
  int l16 = lane & 15, g = lane >> 4;
  int m0 = blockIdx.y * 128 + wid * 16;
  int n0 = blockIdx.x * 64;
  int nk = Kdim / 32;

  v8f acc[4];
#pragma unroll
  for (int j = 0; j < 4; ++j)
#pragma unroll
    for (int i = 0; i < 8; ++i) acc[j][i] = 0.f;

  const unsigned short* xrow  = X + (size_t)(m0 + l16) * Kdim;
  const unsigned short* wbase = W + (size_t)n0 * Kdim;

#if HAVE_TDM
  if (wid == 0)
    tdm_load_tile(wbase, (unsigned)(uintptr_t)&wtile[0][0][0], Kdim, Ndim);
#endif

  for (int it = 0; it < nk; ++it) {
    int kk = it * 32;
    int p  = it & 1;
#if HAVE_TDM
    if (wid == 0) tdm_wait0();          // current buffer landed in LDS
    __syncthreads();                    // release all waves; prior readers done
    if (wid == 0 && it + 1 < nk)
      tdm_load_tile(wbase + (kk + 32),
                    (unsigned)(uintptr_t)&wtile[(it + 1) & 1][0][0], Kdim, Ndim);
#else
    // cooperative fallback: 256 threads copy the 64x32 bf16 tile (one b128 each)
    {
      int r = threadIdx.x >> 2, c = threadIdx.x & 3;
      *(uint4*)&wtile[0][r][c * 8] =
          *(const uint4*)(W + (size_t)(n0 + r) * Kdim + kk + c * 8);
      p = 0;
    }
    __syncthreads();
#endif
    // A fragment: K runs [kk+g*8 .. +7] and [kk+16+g*8 .. +7] (ISA 16-bit A layout)
    v16bf a = frag_from2(xrow + kk + g * 8, xrow + kk + 16 + g * 8);
    __builtin_prefetch(xrow + kk + 64, 0, 1);
#pragma unroll
    for (int j = 0; j < 4; ++j) {
      v16bf bf = frag_from2(&wtile[p][j * 16 + l16][g * 8], &wtile[p][j * 16 + l16][16 + g * 8]);
      acc[j] = WMMA_BF16(a, bf, acc[j]);
    }
#if !HAVE_TDM
    __syncthreads();
#endif
  }

#pragma unroll
  for (int j = 0; j < 4; ++j) {
    int n = n0 + j * 16 + l16;
    float bv = bias ? bias[n] : 0.f;
#pragma unroll
    for (int i = 0; i < 8; ++i) {
      int m = m0 + i + 8 * g;
      float v = acc[j][i] + bv;
      size_t idx = (size_t)m * Ndim + n;
      if (add)  v += add[idx];
      if (relu) v = v > 0.f ? v : 0.f;
      if (outF) outF[idx] = v;
      if (outB) outB[idx] = f2bf(v);
    }
  }
}

// ---------------- flash attention: one wave = 16 queries of one (b,head) ----------------
__global__ void __launch_bounds__(256)
k_attn(const unsigned short* __restrict__ q, const unsigned short* __restrict__ k,
       const unsigned short* __restrict__ vt, const int* __restrict__ amask,
       unsigned short* __restrict__ ctx) {
  __shared__ __align__(16) unsigned short pbuf[8][16][32];  // per-wave P staging
  int w = threadIdx.x >> 5, lane = threadIdx.x & 31;
  int l16 = lane & 15, g = lane >> 4;
  int wg = blockIdx.x * 8 + w;
  int qt   = wg & 31;                 // 32 query tiles of 16
  int head = (wg >> 5) % NH_;
  int b    = wg / (32 * NH_);
  const float scale = 0.125f;         // HD^-0.5

  const unsigned short* qrow = q + ((size_t)(b * L_ + qt * 16 + l16) * H_ + head * HD_);
  v16bf qa0 = frag_from2(qrow + g * 8,      qrow + 16 + g * 8);
  v16bf qa1 = frag_from2(qrow + 32 + g * 8, qrow + 48 + g * 8);

  v8f o[4];
  float mi[8], li[8];
#pragma unroll
  for (int j = 0; j < 4; ++j)
#pragma unroll
    for (int i = 0; i < 8; ++i) o[j][i] = 0.f;
#pragma unroll
  for (int i = 0; i < 8; ++i) { mi[i] = -1e30f; li[i] = 0.f; }

  const unsigned short* kbase = k + ((size_t)(b * L_) * H_ + head * HD_);
  const unsigned short* vbase = vt + (((size_t)b * NH_ + head) * HD_) * L_;
  const int* am = amask + b * L_;

  for (int kt = 0; kt < L_ / 32; ++kt) {
    int key0 = kt * 32;
    v8f s0, s1;
#pragma unroll
    for (int i = 0; i < 8; ++i) { s0[i] = 0.f; s1[i] = 0.f; }
    {
      // B = K^T tiles: lane -> key, halves -> head dims [g*8..+7] and [16+g*8..+7]
      const unsigned short* kr0 = kbase + (size_t)(key0 + l16) * H_;
      const unsigned short* kr1 = kbase + (size_t)(key0 + 16 + l16) * H_;
      s0 = WMMA_BF16(qa0, frag_from2(kr0 + g * 8, kr0 + 16 + g * 8),      s0);
      s0 = WMMA_BF16(qa1, frag_from2(kr0 + 32 + g * 8, kr0 + 48 + g * 8), s0);
      s1 = WMMA_BF16(qa0, frag_from2(kr1 + g * 8, kr1 + 16 + g * 8),      s1);
      s1 = WMMA_BF16(qa1, frag_from2(kr1 + 32 + g * 8, kr1 + 48 + g * 8), s1);
    }
    int am0 = am[key0 + l16];
    int am1 = am[key0 + 16 + l16];
#pragma unroll
    for (int i = 0; i < 8; ++i) {
      float sv0 = am0 ? s0[i] * scale : -1e30f;
      float sv1 = am1 ? s1[i] * scale : -1e30f;
      float rmax = fmaxf(sv0, sv1);
#pragma unroll
      for (int off = 1; off <= 8; off <<= 1) rmax = fmaxf(rmax, __shfl_xor(rmax, off, 32));
      float mnew  = fmaxf(mi[i], rmax);
      float alpha = __expf(mi[i] - mnew);
      float p0 = __expf(sv0 - mnew);
      float p1 = __expf(sv1 - mnew);
      float rsum = p0 + p1;
#pragma unroll
      for (int off = 1; off <= 8; off <<= 1) rsum += __shfl_xor(rsum, off, 32);
      li[i] = li[i] * alpha + rsum;
      mi[i] = mnew;
#pragma unroll
      for (int j = 0; j < 4; ++j) o[j][i] *= alpha;
      int mrow = i + 8 * g;                     // C-layout row held by this lane group
      pbuf[w][mrow][l16]      = f2bf(p0);
      pbuf[w][mrow][16 + l16] = f2bf(p1);
    }
    // LDS is in-order per wave; explicit CDNA5 split-counter wait + compiler barrier.
    asm volatile("s_wait_dscnt 0x0" ::: "memory");
    // Re-read P in A-fragment layout (row = l16, two contiguous 8-half K runs).
    v16bf pa = frag_from2(&pbuf[w][l16][g * 8], &pbuf[w][l16][16 + g * 8]);
#pragma unroll
    for (int j = 0; j < 4; ++j) {
      const unsigned short* vr = vbase + (size_t)(j * 16 + l16) * L_ + key0 + g * 8;
      o[j] = WMMA_BF16(pa, frag_from2(vr, vr + 16), o[j]);
    }
  }

#pragma unroll
  for (int i = 0; i < 8; ++i) {
    float inv = 1.0f / fmaxf(li[i], 1e-20f);
    int m = b * L_ + qt * 16 + i + 8 * g;
#pragma unroll
    for (int j = 0; j < 4; ++j)
      ctx[(size_t)m * H_ + head * HD_ + j * 16 + l16] = f2bf(o[j][i] * inv);
  }
}

// ---------------- masked mean-pool + classifier ----------------
__global__ void k_pool_cls(const float* __restrict__ x, const int* __restrict__ amask,
                           const float* __restrict__ cw, const float* __restrict__ cb,
                           const float* __restrict__ cm, float* __restrict__ out) {
  __shared__ float pooled[H_];
  __shared__ float red[256];
  int b = blockIdx.x, tid = threadIdx.x;
  float cnt = 0.f;
  for (int l = 0; l < L_; ++l) cnt += (float)(amask[b * L_ + l] != 0);
  float inv = 1.f / fmaxf(cnt, 1.f);
  for (int c = tid; c < H_; c += 256) {
    float s = 0.f;
    for (int l = 0; l < L_; ++l)
      if (amask[b * L_ + l]) s += x[(size_t)(b * L_ + l) * H_ + c];
    pooled[c] = s * inv;
  }
  __syncthreads();
  for (int co = 0; co < NC_; ++co) {
    float p = 0.f;
    for (int c = tid; c < H_; c += 256)
      p += pooled[c] * cw[co * H_ + c] * cm[co * H_ + c];
    red[tid] = p;
    __syncthreads();
    for (int off = 128; off; off >>= 1) {
      if (tid < off) red[tid] += red[tid + off];
      __syncthreads();
    }
    if (tid == 0) out[b * NC_ + co] = red[0] + cb[co];
    __syncthreads();
  }
}

// ---------------- host orchestration ----------------
extern "C" void kernel_launch(void* const* d_in, const int* in_sizes, int n_in,
                              void* d_out, int out_size, void* d_ws, size_t ws_size,
                              hipStream_t stream) {
  const int*   ids    = (const int*)d_in[0];
  const int*   amask  = (const int*)d_in[1];
  const float* te     = (const float*)d_in[2];
  const float* pe     = (const float*)d_in[3];
  const float* Wq_w   = (const float*)d_in[4];
  const float* Wq_b   = (const float*)d_in[5];
  const float* Wk_w   = (const float*)d_in[6];
  const float* Wk_b   = (const float*)d_in[7];
  const float* Wv_w   = (const float*)d_in[8];
  const float* Wv_b   = (const float*)d_in[9];
  const float* Wo_w   = (const float*)d_in[10];
  const float* attn_m = (const float*)d_in[11];
  const float* ln1_g  = (const float*)d_in[12];
  const float* ln1_b  = (const float*)d_in[13];
  const float* ln2_g  = (const float*)d_in[14];
  const float* ln2_b  = (const float*)d_in[15];
  const float* ff1_w  = (const float*)d_in[16];
  const float* ff1_b  = (const float*)d_in[17];
  const float* ff_m   = (const float*)d_in[18];
  const float* ff2_w  = (const float*)d_in[19];
  const float* ff2_b  = (const float*)d_in[20];
  const float* skip_w = (const float*)d_in[21];
  const float* skip_m = (const float*)d_in[22];
  const float* cls_w  = (const float*)d_in[23];
  const float* cls_b  = (const float*)d_in[24];
  const float* cls_m  = (const float*)d_in[25];
  (void)in_sizes; (void)n_in; (void)out_size; (void)ws_size;

  char* ws = (char*)d_ws;
  size_t off = 0;
  auto alloc = [&](size_t bytes) -> char* {
    char* p = ws + off;
    off += (bytes + 255) & ~(size_t)255;
    return p;
  };

  float* actsF0 = (float*)alloc((size_t)M_ * H_ * 4);
  float* actsF1 = (float*)alloc((size_t)M_ * H_ * 4);
  unsigned short* actsB[NL_ + 1];
  for (int i = 0; i <= NL_; ++i) actsB[i] = (unsigned short*)alloc((size_t)M_ * H_ * 2);
  unsigned short* lnx  = (unsigned short*)alloc((size_t)M_ * H_ * 2);
  unsigned short* lnh  = (unsigned short*)alloc((size_t)M_ * H_ * 2);
  unsigned short* qAct = (unsigned short*)alloc((size_t)M_ * H_ * 2);
  unsigned short* kAct = (unsigned short*)alloc((size_t)M_ * H_ * 2);
  unsigned short* vAct = (unsigned short*)alloc((size_t)M_ * H_ * 2);
  unsigned short* vtAct= (unsigned short*)alloc((size_t)M_ * H_ * 2);
  unsigned short* ctx  = (unsigned short*)alloc((size_t)M_ * H_ * 2);
  unsigned short* ffAct= (unsigned short*)alloc((size_t)M_ * FF_ * 2);
  unsigned short* wq   = (unsigned short*)alloc((size_t)NL_ * H_ * H_ * 2);
  unsigned short* wk   = (unsigned short*)alloc((size_t)NL_ * H_ * H_ * 2);
  unsigned short* wv   = (unsigned short*)alloc((size_t)NL_ * H_ * H_ * 2);
  unsigned short* wo   = (unsigned short*)alloc((size_t)NL_ * H_ * H_ * 2);
  unsigned short* wf1  = (unsigned short*)alloc((size_t)NL_ * FF_ * H_ * 2);
  unsigned short* wf2  = (unsigned short*)alloc((size_t)NL_ * H_ * FF_ * 2);
  unsigned short* wsk  = (unsigned short*)alloc((size_t)NSKIP_ * H_ * H_ * 2);

  dim3 blk(256);
  const int nHH = NL_ * H_ * H_;
  const int nFH = NL_ * FF_ * H_;
  const int nSK = NSKIP_ * H_ * H_;
  k_wprep<<<(nHH + 255) / 256, blk, 0, stream>>>(Wq_w, nullptr, wq, nHH);
  k_wprep<<<(nHH + 255) / 256, blk, 0, stream>>>(Wk_w, nullptr, wk, nHH);
  k_wprep<<<(nHH + 255) / 256, blk, 0, stream>>>(Wv_w, nullptr, wv, nHH);
  k_wprep<<<(nHH + 255) / 256, blk, 0, stream>>>(Wo_w, attn_m, wo, nHH);
  k_wprep<<<(nFH + 255) / 256, blk, 0, stream>>>(ff1_w, ff_m, wf1, nFH);
  k_wprep<<<(nFH + 255) / 256, blk, 0, stream>>>(ff2_w, nullptr, wf2, nFH);
  k_wprep<<<(nSK + 255) / 256, blk, 0, stream>>>(skip_w, skip_m, wsk, nSK);

  k_embed<<<(M_ * H_) / 256, blk, 0, stream>>>(ids, te, pe, actsF0, actsB[0]);

  float* actsF[2] = {actsF0, actsF1};
  dim3 gH(H_ / 64, M_ / 128);
  dim3 gF(FF_ / 64, M_ / 128);
  for (int i = 0; i < NL_; ++i) {
    float* res = actsF[i & 1];
    float* h   = actsF[(i + 1) & 1];
    k_ln<<<M_ / 8, blk, 0, stream>>>(res, ln1_g + i * H_, ln1_b + i * H_, lnx);
    k_gemm<<<gH, blk, 0, stream>>>(lnx, wq + (size_t)i * H_ * H_, Wq_b + i * H_,
                                   nullptr, nullptr, qAct, M_, H_, H_, 0);
    k_gemm<<<gH, blk, 0, stream>>>(lnx, wk + (size_t)i * H_ * H_, Wk_b + i * H_,
                                   nullptr, nullptr, kAct, M_, H_, H_, 0);
    k_gemm<<<gH, blk, 0, stream>>>(lnx, wv + (size_t)i * H_ * H_, Wv_b + i * H_,
                                   nullptr, nullptr, vAct, M_, H_, H_, 0);
    k_transpose_v<<<(M_ * H_) / 256, blk, 0, stream>>>(vAct, vtAct);
    k_attn<<<(B_ * NH_ * (L_ / 16)) / 8, blk, 0, stream>>>(qAct, kAct, vtAct, amask, ctx);
    k_gemm<<<gH, blk, 0, stream>>>(ctx, wo + (size_t)i * H_ * H_, nullptr,
                                   res, h, nullptr, M_, H_, H_, 0);
    k_ln<<<M_ / 8, blk, 0, stream>>>(h, ln2_g + i * H_, ln2_b + i * H_, lnh);
    k_gemm<<<gF, blk, 0, stream>>>(lnh, wf1 + (size_t)i * FF_ * H_, ff1_b + i * FF_,
                                   nullptr, nullptr, ffAct, M_, FF_, H_, 1);
    k_gemm<<<gH, blk, 0, stream>>>(ffAct, wf2 + (size_t)i * H_ * FF_, ff2_b + i * H_,
                                   h, h, nullptr, M_, H_, FF_, 0);
    int t = i + 1, base = (t - 1) * (t - 2) / 2;
    for (int s = 0; s <= t - 2; ++s) {
      int j = base + s;  // matches reference SKIP_PAIRS enumeration order
      k_gemm<<<gH, blk, 0, stream>>>(actsB[s], wsk + (size_t)j * H_ * H_, nullptr,
                                     h, h, nullptr, M_, H_, H_, 0);
    }
    k_f2bf<<<(M_ * H_) / 256, blk, 0, stream>>>(h, actsB[i + 1], M_ * H_);
  }

  k_pool_cls<<<B_, blk, 0, stream>>>(actsF[NL_ & 1], amask, cls_w, cls_b, cls_m,
                                     (float*)d_out);
}
